// BiMambaWrapper_10849087389982
// MI455X (gfx1250) — compile-verified, matrix operand order checked by reading, not run on hardware
//
#include <hip/hip_runtime.h>
#include <cstdint>
#include <cstddef>

// ---------------- problem constants ----------------
#define DM   512      // D_MODEL
#define DI   1024     // D_INNER
#define DS   16       // D_STATE
#define DTR  32       // DT_RANK
#define NB   4        // batch
#define TT   2048     // seq len
#define BT   (NB*TT)  // 8192 rows

typedef __bf16 bf16;
typedef __attribute__((ext_vector_type(16))) __bf16 v16bf;
typedef __attribute__((ext_vector_type(8)))  float  v8f;

__device__ __forceinline__ float dsigmoid(float x) { return 1.0f / (1.0f + __expf(-x)); }

// CDNA5 async global->LDS 16B copy (ASYNCcnt-tracked, bypasses VGPRs).
// ldsoff: LDS byte address (flat LDS-aperture address truncates to addr[31:0], ISA 10.2)
// gaddr : 64-bit global address
__device__ __forceinline__ void async_copy_b128(unsigned int ldsoff, unsigned long long gaddr) {
    asm volatile("global_load_async_to_lds_b128 %0, %1, off"
                 :: "v"(ldsoff), "v"(gaddr) : "memory");
}
__device__ __forceinline__ void async_wait0() {
    asm volatile("s_wait_asynccnt 0" ::: "memory");
}

// ---------------- prep kernels ----------------
__global__ __launch_bounds__(256) void f32_to_bf16_kernel(const float* __restrict__ in,
                                                          bf16* __restrict__ out, int n) {
    int i = blockIdx.x * 256 + threadIdx.x;
    if (i < n) out[i] = (bf16)in[i];
}

// x (B, C=512, T) f32 -> Xf[(b*T+t)*512 + c], Xb[(b*T+(T-1-t))*512 + c] in bf16
__global__ __launch_bounds__(256) void xprep_kernel(const float* __restrict__ x,
                                                    bf16* __restrict__ Xf,
                                                    bf16* __restrict__ Xb) {
    int i = blockIdx.x * 256 + threadIdx.x;            // over B*512*T
    if (i >= NB * DM * TT) return;
    int t = i % TT;
    int c = (i / TT) % DM;
    int b = i / (DM * TT);
    bf16 v = (bf16)x[i];
    Xf[((size_t)(b * TT + t)) * DM + c] = v;
    Xb[((size_t)(b * TT + (TT - 1 - t))) * DM + c] = v;
}

// ---------------- WMMA GEMM: C[M,N] = A[M,K] * B[N,K]^T ----------------
// A, B bf16 row-major; C is TOut with leading dim ldC and column offset colOff.
// Tiles: block BMxBN, K-step BK=64; 8 waves, each wave computes (WM*16)x(WN*16).
// Staging uses CDNA5 async global->LDS DMA (no VGPR round-trip in the hot loop).
template <int BM, int BN, int BK, int WM, int WN, typename TOut>
__global__ __launch_bounds__(256) void gemm_bf16_wmma(const bf16* __restrict__ A,
                                                      const bf16* __restrict__ Bw,
                                                      TOut* __restrict__ C,
                                                      int M, int N, int K,
                                                      int ldC, int colOff) {
    constexpr int WAVES_M = BM / (16 * WM);
    constexpr int WAVES_N = BN / (16 * WN);
    static_assert(WAVES_M * WAVES_N == 8, "8 waves per block");
    constexpr int LDP = BK + 8;          // padded LDS stride (elements) -> 16B aligned rows
    constexpr int CHA = BM * (BK / 8);   // 16-byte chunks for A tile
    constexpr int CHB = BN * (BK / 8);

    __shared__ __align__(16) bf16 sA[BM * LDP];
    __shared__ __align__(16) bf16 sB[BN * LDP];

    const int tid   = threadIdx.x;
    const int lane  = tid & 31;
    const int wave  = tid >> 5;
    const int wm0   = (wave % WAVES_M) * WM * 16;
    const int wn0   = (wave / WAVES_M) * WN * 16;
    const int bM    = blockIdx.y * BM;
    const int bN    = blockIdx.x * BN;

    const int fm    = lane & 15;   // row within 16x16 frag
    const int fh    = lane >> 4;   // K-half selector

    v8f acc[WM][WN];
    #pragma unroll
    for (int i = 0; i < WM; ++i)
        #pragma unroll
        for (int j = 0; j < WN; ++j) acc[i][j] = (v8f){0,0,0,0,0,0,0,0};

    for (int k0 = 0; k0 < K; k0 += BK) {
        // ---- stage tiles into LDS via async DMA (ASYNCcnt) ----
        #pragma unroll 2
        for (int i = tid; i < CHA; i += 256) {
            int r = i >> 3, c = (i & 7) * 8;
            async_copy_b128((unsigned int)(uintptr_t)(&sA[r * LDP + c]),
                            (unsigned long long)(uintptr_t)(A + (size_t)(bM + r) * K + k0 + c));
        }
        #pragma unroll 2
        for (int i = tid; i < CHB; i += 256) {
            int r = i >> 3, c = (i & 7) * 8;
            async_copy_b128((unsigned int)(uintptr_t)(&sB[r * LDP + c]),
                            (unsigned long long)(uintptr_t)(Bw + (size_t)(bN + r) * K + k0 + c));
        }
        // prefetch next K tile into L2 while we compute (global_prefetch_b8)
        if (k0 + BK < K) {
            __builtin_prefetch(A + (size_t)(bM + (tid & (BM - 1))) * K + k0 + BK, 0, 1);
            __builtin_prefetch(Bw + (size_t)(bN + (tid & (BN - 1))) * K + k0 + BK, 0, 1);
        }
        async_wait0();       // ASYNCcnt not covered by __syncthreads' implicit waits
        __syncthreads();

        // ---- WMMA over BK in steps of 32 ----
        #pragma unroll
        for (int ks = 0; ks < BK / 32; ++ks) {
            v16bf fa[WM], fb[WN];
            #pragma unroll
            for (int i = 0; i < WM; ++i) {
                const bf16* p = &sA[(wm0 + i * 16 + fm) * LDP + ks * 32 + fh * 8];
                union { uint4 u[2]; v16bf v; } t;
                t.u[0] = *(const uint4*)p;
                t.u[1] = *(const uint4*)(p + 16);
                fa[i] = t.v;
            }
            #pragma unroll
            for (int j = 0; j < WN; ++j) {
                const bf16* p = &sB[(wn0 + j * 16 + fm) * LDP + ks * 32 + fh * 8];
                union { uint4 u[2]; v16bf v; } t;
                t.u[0] = *(const uint4*)p;
                t.u[1] = *(const uint4*)(p + 16);
                fb[j] = t.v;
            }
            #pragma unroll
            for (int i = 0; i < WM; ++i)
                #pragma unroll
                for (int j = 0; j < WN; ++j)
                    acc[i][j] = __builtin_amdgcn_wmma_f32_16x16x32_bf16(
                        false, fa[i], false, fb[j], (short)0, acc[i][j], false, false);
        }
        __syncthreads();
    }

    // ---- epilogue: C/D layout -> lanes 0-15 hold M=r (VGPR r), lanes 16-31 hold M=r+8
    const int cn = lane & 15;
    const int ch = lane >> 4;
    #pragma unroll
    for (int i = 0; i < WM; ++i)
        #pragma unroll
        for (int j = 0; j < WN; ++j) {
            int gm = bM + wm0 + i * 16 + ch * 8;
            int gn = bN + wn0 + j * 16 + cn + colOff;
            #pragma unroll
            for (int r = 0; r < 8; ++r)
                C[(size_t)(gm + r) * ldC + gn] = (TOut)acc[i][j][r];
        }
}

// ---------------- depthwise causal conv(k=4) + SiLU ----------------
// xz (BT,2048) f32, xc = cols [0,1024). Writes u f32 and u bf16.
__global__ __launch_bounds__(256) void dwconv_silu_kernel(const float* __restrict__ xz,
                                                          const float* __restrict__ w,   // (DI,1,4)
                                                          const float* __restrict__ bias,
                                                          float* __restrict__ u32,
                                                          bf16* __restrict__ u16) {
    int i = blockIdx.x * 256 + threadIdx.x;  // over BT*DI
    if (i >= BT * DI) return;
    int d  = i & (DI - 1);
    int bt = i >> 10;
    int t  = bt & (TT - 1);
    int b  = bt / TT;
    float acc = bias[d];
    #pragma unroll
    for (int j = 0; j < 4; ++j) {
        int tt = t - 3 + j;
        if (tt >= 0)
            acc = fmaf(w[d * 4 + j], xz[((size_t)(b * TT + tt)) * (2 * DI) + d], acc);
    }
    float s = acc * dsigmoid(acc);
    u32[i] = s;
    u16[i] = (bf16)s;
}

// ---------------- selective scan (fused dt / softplus / D skip / SiLU(z) gate) --------
// one lane per (b,d); h[16], A[16], Wdt[32] register-resident; writes gated bf16 ym.
__global__ __launch_bounds__(256) void scan_kernel(const float* __restrict__ u,     // (BT,DI)
                                                   const float* __restrict__ xdbl,  // (BT,64)
                                                   const float* __restrict__ xz,    // (BT,2048) for z
                                                   const float* __restrict__ Wdt,   // (DI,32)
                                                   const float* __restrict__ bdt,   // (DI)
                                                   const float* __restrict__ Alog,  // (DI,16)
                                                   const float* __restrict__ Dv,    // (DI)
                                                   bf16* __restrict__ ym) {         // (BT,DI)
    int gid = blockIdx.x * 256 + threadIdx.x;  // 0 .. NB*DI-1
    if (gid >= NB * DI) return;
    int d = gid & (DI - 1);
    int b = gid >> 10;

    float wdt[DTR];
    #pragma unroll
    for (int n = 0; n < DTR; ++n) wdt[n] = Wdt[d * DTR + n];
    float Ar[DS];
    #pragma unroll
    for (int n = 0; n < DS; ++n) Ar[n] = -__expf(Alog[d * DS + n]);
    float h[DS];
    #pragma unroll
    for (int n = 0; n < DS; ++n) h[n] = 0.0f;

    const float bd = bdt[d];
    const float Dd = Dv[d];

    for (int t = 0; t < TT; ++t) {
        size_t row = (size_t)(b * TT + t);
        const float* xr = xdbl + row * 64;
        float dt = bd;
        #pragma unroll
        for (int n = 0; n < DTR; ++n) dt = fmaf(wdt[n], xr[n], dt);
        dt = (dt > 20.0f) ? dt : log1pf(__expf(dt));      // softplus

        float ut = u[row * DI + d];
        float du = dt * ut;
        float y  = 0.0f;
        #pragma unroll
        for (int n = 0; n < DS; ++n) {
            float dA = __expf(dt * Ar[n]);
            h[n] = fmaf(h[n], dA, du * xr[DTR + n]);      // B at cols 32..47
            y    = fmaf(h[n], xr[DTR + DS + n], y);       // C at cols 48..63
        }
        y = fmaf(ut, Dd, y);
        float z = xz[row * (2 * DI) + DI + d];
        y *= z * dsigmoid(z);
        ym[row * DI + d] = (bf16)y;
    }
}

// ---------------- GLU + per-block partial sums (deterministic norm, pass 1) ----------
__global__ __launch_bounds__(256) void glu_kernel(const float* __restrict__ y2,   // (BT,1024)
                                                  const float* __restrict__ cb,   // (1024)
                                                  float* __restrict__ g,          // (BT,512)
                                                  float* __restrict__ psum,       // (NB*4096)
                                                  float* __restrict__ psq) {
    int i = blockIdx.x * 256 + threadIdx.x;  // over BT*512 (blocks never straddle a batch)
    int c  = i % DM;
    size_t bt = (size_t)i / DM;
    float a    = y2[bt * DI + c] + cb[c];
    float gate = y2[bt * DI + DM + c] + cb[DM + c];
    float v = a * dsigmoid(gate);
    g[i] = v;

    __shared__ float s1[256], s2[256];
    int tid = threadIdx.x;
    s1[tid] = v; s2[tid] = v * v;
    __syncthreads();
    #pragma unroll
    for (int s = 128; s > 0; s >>= 1) {
        if (tid < s) { s1[tid] += s1[tid + s]; s2[tid] += s2[tid + s]; }
        __syncthreads();
    }
    if (tid == 0) { psum[blockIdx.x] = s1[0]; psq[blockIdx.x] = s2[0]; }
}

// pass 2: one block per batch reduces its 4096 partials (fixed order -> deterministic)
__global__ __launch_bounds__(256) void stats_kernel(const float* __restrict__ psum,
                                                    const float* __restrict__ psq,
                                                    float* __restrict__ stats) {   // (NB,2)
    const int b = blockIdx.x;
    const int per = (TT * DM) / 256;  // 4096 blocks per batch
    float a = 0.0f, q = 0.0f;
    for (int i = threadIdx.x; i < per; i += 256) {
        a += psum[b * per + i];
        q += psq[b * per + i];
    }
    __shared__ float s1[256], s2[256];
    s1[threadIdx.x] = a; s2[threadIdx.x] = q;
    __syncthreads();
    #pragma unroll
    for (int s = 128; s > 0; s >>= 1) {
        if (threadIdx.x < s) { s1[threadIdx.x] += s1[threadIdx.x + s];
                               s2[threadIdx.x] += s2[threadIdx.x + s]; }
        __syncthreads();
    }
    if (threadIdx.x == 0) { stats[b * 2] = s1[0]; stats[b * 2 + 1] = s2[0]; }
}

// ---------------- normalize + affine + transpose to (B,C,T) ----------------
__global__ __launch_bounds__(256) void norm_out_kernel(const float* __restrict__ g,      // (BT,512)
                                                       const float* __restrict__ stats,  // (NB,2)
                                                       const float* __restrict__ gw,
                                                       const float* __restrict__ gb,
                                                       float* __restrict__ out) {        // (B,512,T)
    int i = blockIdx.x * 256 + threadIdx.x;  // over B*512*T
    if (i >= NB * DM * TT) return;
    int t = i % TT;
    int c = (i / TT) % DM;
    int b = i / (DM * TT);
    const float cnt = (float)(DM * TT);
    float mu  = stats[b * 2] / cnt;
    float var = stats[b * 2 + 1] / cnt - mu * mu;
    float v = g[((size_t)(b * TT + t)) * DM + c];
    out[i] = (v - mu) * rsqrtf(var + 1e-5f) * gw[c] + gb[c];
}

// ---------------- host-side launch ----------------
extern "C" void kernel_launch(void* const* d_in, const int* in_sizes, int n_in,
                              void* d_out, int out_size, void* d_ws, size_t ws_size,
                              hipStream_t stream) {
    (void)in_sizes; (void)n_in; (void)out_size; (void)ws_size;
    const float* x      = (const float*)d_in[0];
    const float* conv_w = (const float*)d_in[1];
    const float* conv_b = (const float*)d_in[2];
    const float* gn_w   = (const float*)d_in[3];
    const float* gn_b   = (const float*)d_in[4];
    const float* Win[2]  = {(const float*)d_in[5],  (const float*)d_in[14]};
    const float* convw[2]= {(const float*)d_in[6],  (const float*)d_in[15]};
    const float* convb[2]= {(const float*)d_in[7],  (const float*)d_in[16]};
    const float* Wx[2]   = {(const float*)d_in[8],  (const float*)d_in[17]};
    const float* Wdt[2]  = {(const float*)d_in[9],  (const float*)d_in[18]};
    const float* bdt[2]  = {(const float*)d_in[10], (const float*)d_in[19]};
    const float* Alog[2] = {(const float*)d_in[11], (const float*)d_in[20]};
    const float* Dv[2]   = {(const float*)d_in[12], (const float*)d_in[21]};
    const float* Wout[2] = {(const float*)d_in[13], (const float*)d_in[22]};
    float* out = (float*)d_out;

    // -------- workspace carve-up --------
    char* wbase = (char*)d_ws;
    size_t off = 0;
    auto take = [&](size_t bytes) -> void* {
        void* p = wbase + off;
        off = (off + bytes + 255) & ~(size_t)255;
        return p;
    };
    bf16* Xd[2];
    Xd[0] = (bf16*)take((size_t)BT * DM * 2);
    Xd[1] = (bf16*)take((size_t)BT * DM * 2);
    bf16 *Win_bf[2], *Wx_bf[2], *Wout_bf[2];
    for (int d = 0; d < 2; ++d) Win_bf[d]  = (bf16*)take((size_t)2 * DI * DM * 2);
    for (int d = 0; d < 2; ++d) Wx_bf[d]   = (bf16*)take((size_t)(DTR + 2 * DS) * DI * 2);
    for (int d = 0; d < 2; ++d) Wout_bf[d] = (bf16*)take((size_t)DM * DI * 2);
    bf16* Wpw_bf = (bf16*)take((size_t)DI * DI * 2);
    float *xz[2], *u32[2], *xdbl[2];
    bf16 *u16[2], *ym[2];
    for (int d = 0; d < 2; ++d) xz[d]   = (float*)take((size_t)BT * 2 * DI * 4);
    for (int d = 0; d < 2; ++d) u32[d]  = (float*)take((size_t)BT * DI * 4);
    for (int d = 0; d < 2; ++d) u16[d]  = (bf16*)take((size_t)BT * DI * 2);
    for (int d = 0; d < 2; ++d) xdbl[d] = (float*)take((size_t)BT * 64 * 4);
    for (int d = 0; d < 2; ++d) ym[d]   = (bf16*)take((size_t)BT * DI * 2);
    bf16*  ycat  = (bf16*)take((size_t)BT * DI * 2);
    float* y2    = (float*)take((size_t)BT * DI * 4);
    float* g     = (float*)take((size_t)BT * DM * 4);
    float* psum  = (float*)take((size_t)NB * 4096 * 4);
    float* psq   = (float*)take((size_t)NB * 4096 * 4);
    float* stats = (float*)take((size_t)NB * 2 * 4);

    // -------- 1. input transpose/convert + weight conversion --------
    xprep_kernel<<<(NB * DM * TT + 255) / 256, 256, 0, stream>>>(x, Xd[0], Xd[1]);
    auto cvt = [&](const float* src, bf16* dst, int n) {
        f32_to_bf16_kernel<<<(n + 255) / 256, 256, 0, stream>>>(src, dst, n);
    };
    for (int d = 0; d < 2; ++d) {
        cvt(Win[d],  Win_bf[d],  2 * DI * DM);
        cvt(Wx[d],   Wx_bf[d],   (DTR + 2 * DS) * DI);
        cvt(Wout[d], Wout_bf[d], DM * DI);
    }
    cvt(conv_w, Wpw_bf, DI * DI);  // (1024,1024,1) == (N,K) row-major

    // -------- per-direction Mamba pipeline --------
    for (int d = 0; d < 2; ++d) {
        // xz = X @ Win^T : M=8192, N=2048, K=512
        gemm_bf16_wmma<128, 128, 64, 2, 4, float>
            <<<dim3(2 * DI / 128, BT / 128), 256, 0, stream>>>(
                Xd[d], Win_bf[d], xz[d], BT, 2 * DI, DM, 2 * DI, 0);
        // depthwise conv + SiLU
        dwconv_silu_kernel<<<(BT * DI + 255) / 256, 256, 0, stream>>>(
            xz[d], convw[d], convb[d], u32[d], u16[d]);
        // x_dbl = u @ Wx^T : M=8192, N=64, K=1024
        gemm_bf16_wmma<256, 64, 64, 2, 4, float>
            <<<dim3(1, BT / 256), 256, 0, stream>>>(
                u16[d], Wx_bf[d], xdbl[d], BT, 64, DI, 64, 0);
        // selective scan (fused dt/softplus/D/SiLU(z))
        scan_kernel<<<(NB * DI + 255) / 256, 256, 0, stream>>>(
            u32[d], xdbl[d], xz[d], Wdt[d], bdt[d], Alog[d], Dv[d], ym[d]);
        // out proj into concat buffer: M=8192, N=512, K=1024, colOff = d*512
        gemm_bf16_wmma<128, 128, 64, 2, 4, bf16>
            <<<dim3(DM / 128, BT / 128), 256, 0, stream>>>(
                ym[d], Wout_bf[d], ycat, BT, DM, DI, DI, d * DM);
    }

    // -------- pointwise mixing conv: y2 = ycat @ conv_w^T --------
    gemm_bf16_wmma<128, 128, 64, 2, 4, float>
        <<<dim3(DI / 128, BT / 128), 256, 0, stream>>>(
            ycat, Wpw_bf, y2, BT, DI, DI, DI, 0);

    // -------- GLU + deterministic instance-norm stats + output --------
    glu_kernel<<<(BT * DM) / 256, 256, 0, stream>>>(y2, conv_b, g, psum, psq);
    stats_kernel<<<NB, 256, 0, stream>>>(psum, psq, stats);
    norm_out_kernel<<<(NB * DM * TT + 255) / 256, 256, 0, stream>>>(
        g, stats, gn_w, gn_b, out);
}
